// FeedForwardQuantum_65481071397339
// MI455X (gfx1250) — matrix operand order repaired and checked by reading, not compiled, not run on hardware
//
#include <hip/hip_runtime.h>

typedef __attribute__((ext_vector_type(16))) __bf16 v16bf;
typedef __attribute__((ext_vector_type(8)))  float  v8f;

namespace {
constexpr int MTOK = 8 * 4096;   // B*S tokens
constexpr int NQ   = 8;
constexpr int FFN  = 2048;
constexpr int EMB  = 512;

constexpr int TM = 128;          // token tile per block
constexpr int TN = 128;          // emb tile per block
constexpr int TK = 32;           // ffn chunk = WMMA K
constexpr int SH = TK + 8;       // LDS row stride in bf16 (80B: 16B-aligned, padded)
constexpr int NCH = FFN / TK;    // 64 chunks

union FragU { uint4 u[2]; v16bf v; };
}

__global__ __launch_bounds__(256)
void ffq_fused(const float* __restrict__ x, const float* __restrict__ theta,
               const float* __restrict__ W1, const float* __restrict__ b1,
               const float* __restrict__ W2, const float* __restrict__ b2,
               float* __restrict__ out)
{
  // double-buffered tiles: 4 x 10 KB = 40 KB LDS total
  __shared__ __bf16 s_h  [2][TM * SH];   // h chunk, row-major [m][k]
  __shared__ __bf16 s_w2t[2][TN * SH];   // W2 chunk, transposed [n][k]

  const int t    = threadIdx.x;
  const int lane = t & 31;
  const int half = lane >> 4;            // 0: lanes 0-15, 1: lanes 16-31
  const int r16  = lane & 15;
  const int wv   = __builtin_amdgcn_readfirstlane(t >> 5);  // wave id (uniform)
  const int wm   = wv & 3;               // 4 M-groups of 32 rows
  const int wn   = wv >> 2;              // 2 N-groups of 64 cols
  const int cg   = __builtin_amdgcn_readfirstlane(t >> 6);  // wave-uniform col group

  const int m0 = blockIdx.x * TM;        // token base
  const int e0 = blockIdx.y * TN;        // emb base

  // ---- per-thread resident q: 2 rows x 8 wires ----
  const int rp = t & 63;
  const int r0 = 2 * rp;
  float q0[NQ], q1[NQ];
  #pragma unroll
  for (int i = 0; i < NQ; ++i) {
    const float ct = cosf(theta[i]);
    q0[i] = cosf(x[(m0 + r0    ) * NQ + i]) * ct;
    q1[i] = cosf(x[(m0 + r0 + 1) * NQ + i]) * ct;
  }

  const int n_w2  = t & 127;             // W2 staging column
  const int kb_w2 = (t >> 7) * 16;       // 0 or 16

  // producer: stage chunk j (h tile + transposed W2 tile) into buffer p.
  // W1/b1 indices are wave-uniform -> SMEM scalar loads, no LDS needed.
  auto produce = [&](int j, int p) {
    const int k0 = j * TK;

    alignas(16) __bf16 wbuf[16];
    #pragma unroll
    for (int jj = 0; jj < 16; ++jj)
      wbuf[jj] = (__bf16)W2[(k0 + kb_w2 + jj) * EMB + e0 + n_w2];
    *(uint4*)&s_w2t[p][n_w2 * SH + kb_w2]     = *(uint4*)&wbuf[0];
    *(uint4*)&s_w2t[p][n_w2 * SH + kb_w2 + 8] = *(uint4*)&wbuf[8];

    alignas(16) __bf16 h0[8], h1[8];
    #pragma unroll
    for (int c = 0; c < 8; ++c) {
      const int k = k0 + cg * 8 + c;     // wave-uniform
      const float bias = b1[k];          // scalar load
      float a0 = bias, a1 = bias;
      #pragma unroll
      for (int i = 0; i < NQ; ++i) {
        const float w = W1[i * FFN + k]; // scalar load, SGPR operand to FMA
        a0 = fmaf(q0[i], w, a0);
        a1 = fmaf(q1[i], w, a1);
      }
      h0[c] = (__bf16)fmaxf(a0, 0.f);
      h1[c] = (__bf16)fmaxf(a1, 0.f);
    }
    *(uint4*)&s_h[p][ r0      * SH + cg * 8] = *(uint4*)&h0[0];
    *(uint4*)&s_h[p][(r0 + 1) * SH + cg * 8] = *(uint4*)&h1[0];
  };

  // ---- accumulators: 2 (M) x 4 (N) tiles of 16x16 f32 per wave ----
  const v8f vzero = {0.f, 0.f, 0.f, 0.f, 0.f, 0.f, 0.f, 0.f};
  v8f acc[2][4];
  #pragma unroll
  for (int mi = 0; mi < 2; ++mi)
    #pragma unroll
    for (int ni = 0; ni < 4; ++ni) acc[mi][ni] = vzero;

  produce(0, 0);
  __syncthreads();

  // one barrier per chunk; next chunk's VALU production co-executes with WMMA
  #pragma unroll 2
  for (int it = 0; it < NCH; ++it) {
    const int cur = it & 1;
    if (it + 1 < NCH) produce(it + 1, cur ^ 1);   // block-uniform branch

    // A (16x32 bf16): lanes 0-15 row M=r16 hold K 0-7 & 16-23; lanes 16-31 hold K 8-15 & 24-31
    FragU a[2];
    #pragma unroll
    for (int mi = 0; mi < 2; ++mi) {
      const __bf16* hrow = &s_h[cur][(wm * 32 + mi * 16 + r16) * SH];
      a[mi].u[0] = *(const uint4*)(hrow + half * 8);
      a[mi].u[1] = *(const uint4*)(hrow + 16 + half * 8);
    }
    // B (32x16 bf16): lanes 0-15 col N=r16 hold K 0-7 & 16-23; lanes 16-31 hold K 8-15 & 24-31
    #pragma unroll
    for (int ni = 0; ni < 4; ++ni) {
      FragU bfr;
      const __bf16* brow = &s_w2t[cur][(wn * 64 + ni * 16 + r16) * SH];
      bfr.u[0] = *(const uint4*)(brow + half * 8);
      bfr.u[1] = *(const uint4*)(brow + 16 + half * 8);
      #pragma unroll
      for (int mi = 0; mi < 2; ++mi)
        acc[mi][ni] = __builtin_amdgcn_wmma_f32_16x16x32_bf16(
            false, a[mi].v, false, bfr.v, (short)0, acc[mi][ni], false, false);
    }
    __syncthreads();
  }

  // ---- epilogue: C layout VGPR j -> row (half*8 + j), col r16 ----
  #pragma unroll
  for (int mi = 0; mi < 2; ++mi) {
    #pragma unroll
    for (int ni = 0; ni < 4; ++ni) {
      const int gm = m0 + wm * 32 + mi * 16 + half * 8;
      const int ge = e0 + wn * 64 + ni * 16 + r16;
      const float bias = b2[ge];
      #pragma unroll
      for (int j = 0; j < 8; ++j)
        out[(gm + j) * EMB + ge] = acc[mi][ni][j] + bias;
    }
  }
}

extern "C" void kernel_launch(void* const* d_in, const int* in_sizes, int n_in,
                              void* d_out, int out_size, void* d_ws, size_t ws_size,
                              hipStream_t stream) {
  const float* x     = (const float*)d_in[0];
  const float* theta = (const float*)d_in[1];
  const float* W1    = (const float*)d_in[2];
  const float* b1    = (const float*)d_in[3];
  const float* W2    = (const float*)d_in[4];
  const float* b2    = (const float*)d_in[5];
  float* out = (float*)d_out;

  dim3 grid(MTOK / TM, EMB / TN);   // (256, 4)
  hipLaunchKernelGGL(ffq_fused, grid, dim3(256), 0, stream,
                     x, theta, W1, b1, W2, b2, out);
}
